// GCN_42417097015695
// MI455X (gfx1250) — matrix operand order repaired and checked by reading, compiled, not run on hardware
//
#include <hip/hip_runtime.h>

typedef __attribute__((ext_vector_type(16))) _Float16 v16h;
typedef __attribute__((ext_vector_type(8)))  float    v8f;

#define N_NODES 50000
#define N_EDGES 1000000
#define FEATS   64

// ---------------- utility kernels ----------------

__global__ void zero_f32(float* __restrict__ p, int n) {
    int i = blockIdx.x * blockDim.x + threadIdx.x;
    if (i < n) p[i] = 0.0f;
}

__global__ void degree_kernel(const int* __restrict__ src, const int* __restrict__ dst,
                              float* __restrict__ deg_out, float* __restrict__ deg_in, int E) {
    int e = blockIdx.x * blockDim.x + threadIdx.x;
    if (e >= E) return;
    atomicAdd(&deg_out[src[e]], 1.0f);
    atomicAdd(&deg_in[dst[e]], 1.0f);
}

// in-place: deg -> rsqrt(clip(deg, 1))
__global__ void norm_kernel(float* __restrict__ p, int n) {
    int i = blockIdx.x * blockDim.x + threadIdx.x;
    if (i >= n) return;
    float v = p[i];
    p[i] = rsqrtf(v < 1.0f ? 1.0f : v);
}

__global__ void efeat_kernel(const float* __restrict__ eh, const float* __restrict__ mu,
                             const float* __restrict__ sigma, float* __restrict__ ef, int E) {
    int e = blockIdx.x * blockDim.x + threadIdx.x;
    if (e >= E) return;
    float d = eh[e] - mu[0];
    ef[e] = logf(d * d / (-sigma[0]));
}

// out[node][f] = in[node][f] * norm[node], float4-vectorized (F=64 -> 16 quads/node)
__global__ void scale_kernel(const float* __restrict__ in, const float* __restrict__ norm,
                             float* __restrict__ out, int n_quads) {
    int idx = blockIdx.x * blockDim.x + threadIdx.x;
    if (idx >= n_quads) return;
    int node = idx >> 4;
    int f4   = (idx & 15) << 2;
    float s = norm[node];
    float4 v = *(const float4*)(in + (size_t)node * FEATS + f4);
    v.x *= s; v.y *= s; v.z *= s; v.w *= s;
    *(float4*)(out + (size_t)node * FEATS + f4) = v;
}

// one thread per (edge, 4 features): gather float4, scale by edge weight, atomic scatter-add
__global__ void scatter_kernel(const float* __restrict__ xs, const float* __restrict__ ef,
                               const int* __restrict__ src, const int* __restrict__ dst,
                               float* __restrict__ agg, int E) {
    int idx = blockIdx.x * blockDim.x + threadIdx.x;
    int e = idx >> 4;
    if (e >= E) return;
    int f4 = (idx & 15) << 2;
    float w = ef[e];
    float4 v = *(const float4*)(xs + (size_t)src[e] * FEATS + f4);
    float* q = agg + (size_t)dst[e] * FEATS + f4;
    atomicAdd(q + 0, v.x * w);
    atomicAdd(q + 1, v.y * w);
    atomicAdd(q + 2, v.z * w);
    atomicAdd(q + 3, v.w * w);
}

// ---------------- WMMA GEMM: out[M,NC] = relu?((A*nd) @ W + b) ----------------
// M = 50000 (3125 exact 16-row tiles), K = 64 (two k-steps of 32), NC in {64,32}.
// Grid: (3125) blocks; blockDim = 32 * (NC/16); wave w handles column tile w.
__global__ void wmma_gemm_kernel(const float* __restrict__ A, const float* __restrict__ nd,
                                 const float* __restrict__ W, const float* __restrict__ bias,
                                 float* __restrict__ out, int NC, int do_relu) {
    const int lane = threadIdx.x & 31;
    const int tn   = threadIdx.x >> 5;      // column tile (wave id within block)
    const int tm   = blockIdx.x;            // row tile

    const int arow  = tm * 16 + (lane & 15);        // row this lane loads for A
    const int col   = tn * 16 + (lane & 15);        // column this lane owns for B/C/D
    const int abase = (lane >> 4) ? 8 : 0;          // A K-offset (ISA 16-bit A layout)
    const int bbase = (lane >> 4) ? 8 : 0;          // B K-offset (same table order as A)

    const float s = nd[arow];
    const float bv = bias[col];

    v8f c;
#pragma unroll
    for (int r = 0; r < 8; ++r) c[r] = bv;

#pragma unroll
    for (int k0 = 0; k0 < 64; k0 += 32) {
        v16h a, b;
        const float* ap = A + (size_t)arow * FEATS + k0 + abase;
#pragma unroll
        for (int j = 0; j < 8; ++j) {
            a[j]     = (_Float16)(ap[j] * s);         // K = k0+abase+j
            a[j + 8] = (_Float16)(ap[16 + j] * s);    // K = k0+abase+16+j
        }
#pragma unroll
        for (int j = 0; j < 8; ++j) {
            b[j]     = (_Float16)W[(size_t)(k0 + bbase + j) * NC + col];
            b[j + 8] = (_Float16)W[(size_t)(k0 + bbase + 16 + j) * NC + col];
        }
        c = __builtin_amdgcn_wmma_f32_16x16x32_f16(
                /*neg_a=*/false, a, /*neg_b=*/false, b,
                /*c_mod=*/(short)0, c, /*reuse_a=*/false, /*reuse_b=*/false);
    }

    const int mbase = (lane >> 4) ? 8 : 0;          // C/D layout: lane half -> M base
#pragma unroll
    for (int r = 0; r < 8; ++r) {
        float v = c[r];
        if (do_relu) v = fmaxf(v, 0.0f);
        out[(size_t)(tm * 16 + mbase + r) * NC + col] = v;
    }
}

// ---------------- pooling ----------------

// one thread per (node, class): atomic add into per-graph sums; class-0 thread counts node
__global__ void pool_kernel(const float* __restrict__ h2, const int* __restrict__ ng,
                            float* __restrict__ gsum, float* __restrict__ gcnt, int N) {
    int idx = blockIdx.x * blockDim.x + threadIdx.x;
    int node = idx >> 5;
    if (node >= N) return;
    int c = idx & 31;
    int g = ng[node];
    atomicAdd(&gsum[g * 32 + c], h2[(size_t)node * 32 + c]);
    if (c == 0) atomicAdd(&gcnt[g], 1.0f);
}

__global__ void finalize_kernel(const float* __restrict__ gsum, const float* __restrict__ gcnt,
                                float* __restrict__ out) {
    int i = blockIdx.x * blockDim.x + threadIdx.x;
    if (i >= 64 * 32) return;
    float cnt = gcnt[i >> 5];
    out[i] = gsum[i] / (cnt < 1.0f ? 1.0f : cnt);
}

// ---------------- launch ----------------

extern "C" void kernel_launch(void* const* d_in, const int* in_sizes, int n_in,
                              void* d_out, int out_size, void* d_ws, size_t ws_size,
                              hipStream_t stream) {
    const float* x     = (const float*)d_in[0];
    const float* eh    = (const float*)d_in[1];
    const int*   src   = (const int*)d_in[2];
    const int*   dst   = (const int*)d_in[3];
    const int*   ng    = (const int*)d_in[4];
    const float* W1    = (const float*)d_in[5];
    const float* b1    = (const float*)d_in[6];
    const float* W2    = (const float*)d_in[7];
    const float* b2    = (const float*)d_in[8];
    const float* mu    = (const float*)d_in[9];
    const float* sigma = (const float*)d_in[10];
    float* out = (float*)d_out;

    const int N = N_NODES, E = N_EDGES;

    // workspace layout (floats)
    float* ws       = (float*)d_ws;
    float* norm_src = ws;                      // N  (deg_out -> rsqrt)
    float* norm_dst = norm_src + N;            // N  (deg_in  -> rsqrt)
    float* efeat    = norm_dst + N;            // E
    float* XS       = efeat + E;               // N*64
    float* AGG      = XS + (size_t)N * FEATS;  // N*64
    float* H        = AGG + (size_t)N * FEATS; // N*64 (layer1 out; reused as layer2 out N*32)
    float* GSUM     = H + (size_t)N * FEATS;   // 64*32
    float* GCNT     = GSUM + 64 * 32;          // 64

    const int T = 256;
    auto blocks = [](long n, int t) { return (int)((n + t - 1) / t); };

    // zero accumulators (every call: deterministic)
    zero_f32<<<blocks(2 * N, T), T, 0, stream>>>(norm_src, 2 * N);
    zero_f32<<<blocks((long)N * FEATS, T), T, 0, stream>>>(AGG, N * FEATS);
    zero_f32<<<blocks(64 * 32 + 64, T), T, 0, stream>>>(GSUM, 64 * 32 + 64);

    // degrees -> norms
    degree_kernel<<<blocks(E, T), T, 0, stream>>>(src, dst, norm_src, norm_dst, E);
    norm_kernel<<<blocks(2 * N, T), T, 0, stream>>>(norm_src, 2 * N);

    // edge features
    efeat_kernel<<<blocks(E, T), T, 0, stream>>>(eh, mu, sigma, efeat, E);

    // ---- layer 1 ----
    scale_kernel<<<blocks((long)N * 16, T), T, 0, stream>>>(x, norm_src, XS, N * 16);
    scatter_kernel<<<blocks((long)E * 16, T), T, 0, stream>>>(XS, efeat, src, dst, AGG, E);
    wmma_gemm_kernel<<<dim3(N / 16), dim3(32 * (64 / 16)), 0, stream>>>(
        AGG, norm_dst, W1, b1, H, 64, 1);

    // ---- layer 2 ----
    scale_kernel<<<blocks((long)N * 16, T), T, 0, stream>>>(H, norm_src, XS, N * 16);
    zero_f32<<<blocks((long)N * FEATS, T), T, 0, stream>>>(AGG, N * FEATS);
    scatter_kernel<<<blocks((long)E * 16, T), T, 0, stream>>>(XS, efeat, src, dst, AGG, E);
    wmma_gemm_kernel<<<dim3(N / 16), dim3(32 * (32 / 16)), 0, stream>>>(
        AGG, norm_dst, W2, b2, H, 32, 0);

    // ---- per-graph mean ----
    pool_kernel<<<blocks((long)N * 32, T), T, 0, stream>>>(H, ng, GSUM, GCNT, N);
    finalize_kernel<<<blocks(64 * 32, T), T, 0, stream>>>(GSUM, GCNT, out);
}
